// Msingi1_61280593379494
// MI455X (gfx1250) — matrix operand order repaired and checked by reading, not compiled, not run on hardware
//
#include <hip/hip_runtime.h>
#include <hip/hip_bf16.h>
#include <math.h>

#define LAYERS 12
#define DMODEL 1024
#define NHEAD  16
#define DHEAD  64
#define FFDIM  4096
#define VOCAB  32000
#define BATCH  2
#define SEQ    1024
#define ROWS   (BATCH * SEQ)   // 2048
#define LN_EPS 1e-5f

typedef _Float16 half_t;
typedef __attribute__((ext_vector_type(4)))  _Float16 v4h;
typedef __attribute__((ext_vector_type(16))) _Float16 v16h;
typedef __attribute__((ext_vector_type(8)))  float    v8f;
typedef __attribute__((ext_vector_type(4)))  int      v4i;

union VH  { v16h v; half_t h[16]; };
union V4H { v4h  v; half_t h[4];  };
union VF  { v8f  v; float  f[8];  };
union F4  { float4 v; float f[4]; };

// gfx1250 async global->LDS builtins (probe-confirmed present; param type is
// address_space(1)/(3) v4i* per clang diagnostic). Guarded for portability.
#if defined(__has_builtin)
#if __has_builtin(__builtin_amdgcn_global_load_async_to_lds_b128) && \
    __has_builtin(__builtin_amdgcn_s_wait_asynccnt)
#define HAVE_ASYNC 1
#endif
#endif
#ifndef HAVE_ASYNC
#define HAVE_ASYNC 0
#endif

#if HAVE_ASYNC
typedef __attribute__((address_space(1))) v4i g_v4i;
typedef __attribute__((address_space(3))) v4i l_v4i;
__device__ inline void async_copy16(const void* g, void* l) {
    __builtin_amdgcn_global_load_async_to_lds_b128((g_v4i*)g, (l_v4i*)l, 0, 0);
}
#endif

__device__ inline v8f zero8() {
    VF u;
#pragma unroll
    for (int i = 0; i < 8; ++i) u.f[i] = 0.0f;
    return u.v;
}

__device__ inline v8f wmma_f16(v16h a, v16h b, v8f c) {
    // D = A(16x32 f16) * B(32x16 f16) + C(16x16 f32)
    return __builtin_amdgcn_wmma_f32_16x16x32_f16(
        false, a, false, b, (short)0, c, false, false);
}

// ---- WMMA fragment loaders (contiguous k order in A and B, wave32) ----------

// A: row-major [M x K] tile at p0=(row0,k0), ld = row stride (elements).
// lane m = lane&15; half h -> K = 16*(lane>=16) + h  (contiguous per lane).
__device__ inline v16h frag_A(const half_t* __restrict__ p0, int ld, int lane) {
    const half_t* p = p0 + (size_t)(lane & 15) * ld + ((lane >> 4) << 4);
    VH a;
#pragma unroll
    for (int i = 0; i < 8; ++i) a.h[i] = p[i];
#pragma unroll
    for (int i = 0; i < 8; ++i) a.h[8 + i] = p[8 + i];
    return a.v;
}

// B from LDS stored transposed [n][k] with row stride LDST halves.
// lane n = lane&15; half h -> K = 16*(lane>=16) + h  (contiguous per lane).
#define LDST 40
__device__ inline v16h frag_B_ldsT(const half_t* p0, int lane) {
    const half_t* p = p0 + (size_t)(lane & 15) * LDST + ((lane >> 4) << 4);
    VH b;
#pragma unroll
    for (int i = 0; i < 16; ++i) b.h[i] = p[i];
    return b.v;
}

// ---- Block reductions (wave32-correct) -------------------------------------

__device__ inline float wave_sum(float v) {
#pragma unroll
    for (int off = 16; off > 0; off >>= 1) v += __shfl_xor(v, off, 32);
    return v;
}
__device__ inline float wave_max(float v) {
#pragma unroll
    for (int off = 16; off > 0; off >>= 1) v = fmaxf(v, __shfl_xor(v, off, 32));
    return v;
}

template <bool MAXOP>
__device__ inline float block_red(float v, float* sm) {
    v = MAXOP ? wave_max(v) : wave_sum(v);
    const int wv = threadIdx.x >> 5;
    const int ln = threadIdx.x & 31;
    const int nw = blockDim.x >> 5;
    if (ln == 0) sm[wv] = v;
    __syncthreads();
    if (wv == 0) {
        float t = (ln < nw) ? sm[ln] : (MAXOP ? -3.0e38f : 0.0f);
        t = MAXOP ? wave_max(t) : wave_sum(t);
        if (ln == 0) sm[0] = t;
    }
    __syncthreads();
    float r = sm[0];
    __syncthreads();
    return r;
}

// ---- GEMM: C[M,N] = act(A_f16[M,K] @ B_f32[K,N] + bias) --------------------
// Block = 16 rows x 256 cols, 4 waves (each wave: 16x64 strip, 4 accumulators).
// Per 32-wide K-step:
//   * B slab 32x256 f32 loaded coalesced (float4), converted f32->f16 in regs,
//     stored transposed [n][k] to LDS as packed ds_store_b64.
//   * A tile 16x32 f16 staged to LDS by wave 0 via async DMA (ASYNCcnt).
//   * Fragments come from LDS: A = 2x16B/lane, B = 2x ds_load_b128/lane.
template <bool GELU, bool OUT_HALF>
__global__ __launch_bounds__(128) void gemm_wmma(
    const half_t* __restrict__ A, const float* __restrict__ B,
    const float* __restrict__ bias, float* __restrict__ Cf,
    half_t* __restrict__ Ch, int M, int N, int K) {
    __shared__ half_t ldsA[16 * LDST];   // 1.25 KB
    __shared__ half_t ldsB[256 * LDST];  // 20 KB, [n][k] transposed, padded
    const int tid  = threadIdx.x;
    const int lane = tid & 31;
    const int wave = tid >> 5;
    const int m0   = blockIdx.y << 4;
    const int nblk = blockIdx.x << 8;    // 256 cols per block
    const int n0w  = wave << 6;          // wave's 64-col slice

    v8f acc0 = zero8(), acc1 = zero8(), acc2 = zero8(), acc3 = zero8();

    for (int k0 = 0; k0 < K; k0 += 32) {
        // --- coalesced load of B slab: unit = 4k x 4n, 4 units per thread ---
        F4 r[4][4];
#pragma unroll
        for (int i = 0; i < 4; ++i) {
            const int u  = tid + (i << 7);   // 0..511
            const int kq = u >> 6;           // 0..7  (k quad)
            const int ng = u & 63;           // 0..63 (n quad)
            const float* bp = B + (size_t)(k0 + (kq << 2)) * N + nblk + (ng << 2);
#pragma unroll
            for (int j = 0; j < 4; ++j)
                r[i][j].v = *(const float4*)(bp + (size_t)j * N);
        }
        __syncthreads();   // previous iteration's LDS consumers are done

        // --- stage A tile (16 x 32 halves): wave 0, 2 x 16B chunks per lane --
        if (wave == 0) {
            const int row = lane >> 2;
            const int off = (lane & 3) << 3;     // half offset, 16B chunk
            const half_t* ga  = A + (size_t)(m0 + row) * K + k0 + off;
            const half_t* ga2 = ga + (size_t)8 * K;
            half_t* la  = &ldsA[row * LDST + off];
            half_t* la2 = la + 8 * LDST;
#if HAVE_ASYNC
            async_copy16(ga, la);
            async_copy16(ga2, la2);
#else
            *(uint4*)la  = *(const uint4*)ga;
            *(uint4*)la2 = *(const uint4*)ga2;
#endif
        }

        // --- store B slab to LDS transposed, packed 4 halves (b64) ----------
#pragma unroll
        for (int i = 0; i < 4; ++i) {
            const int u  = tid + (i << 7);
            const int kq = u >> 6;
            const int ng = u & 63;
#pragma unroll
            for (int j2 = 0; j2 < 4; ++j2) {
                V4H hv;
#pragma unroll
                for (int j = 0; j < 4; ++j) hv.h[j] = (half_t)r[i][j].f[j2];
                *(v4h*)&ldsB[((ng << 2) + j2) * LDST + (kq << 2)] = hv.v;
            }
        }
#if HAVE_ASYNC
        if (wave == 0) __builtin_amdgcn_s_wait_asynccnt(0);
#endif
        __syncthreads();

        // --- fragments + 4 WMMA ---------------------------------------------
        v16h a  = frag_A(ldsA, LDST, lane);
        v16h b0 = frag_B_ldsT(&ldsB[(n0w +  0) * LDST], lane);
        v16h b1 = frag_B_ldsT(&ldsB[(n0w + 16) * LDST], lane);
        v16h b2 = frag_B_ldsT(&ldsB[(n0w + 32) * LDST], lane);
        v16h b3 = frag_B_ldsT(&ldsB[(n0w + 48) * LDST], lane);
        acc0 = wmma_f16(a, b0, acc0);
        acc1 = wmma_f16(a, b1, acc1);
        acc2 = wmma_f16(a, b2, acc2);
        acc3 = wmma_f16(a, b3, acc3);
    }

    const int n  = lane & 15;
    const int rb = (lane >> 4) << 3;
    VF cc[4];
    cc[0].v = acc0; cc[1].v = acc1; cc[2].v = acc2; cc[3].v = acc3;
#pragma unroll
    for (int t = 0; t < 4; ++t) {
        const int col = nblk + n0w + t * 16 + n;
        const float bv = bias ? bias[col] : 0.0f;
#pragma unroll
        for (int r2 = 0; r2 < 8; ++r2) {
            const int row = m0 + rb + r2;
            float x = cc[t].f[r2] + bv;
            if (GELU) x = 0.5f * x * (1.0f + erff(x * 0.70710678118f));
            if (OUT_HALF) Ch[(size_t)row * N + col] = (half_t)x;
            else          Cf[(size_t)row * N + col] = x;
        }
    }
}

// ---- Embedding + LayerNorm0 ------------------------------------------------
__global__ __launch_bounds__(256) void embed_ln_kernel(
    const int* __restrict__ ids, const float* __restrict__ tok,
    const float* __restrict__ pos, const float* __restrict__ g,
    const float* __restrict__ bt, float* __restrict__ h32,
    half_t* __restrict__ h16) {
    __shared__ float sm[8];
    const int row = blockIdx.x;
    const int s = row & (SEQ - 1);
    const int id = ids[row];
    const size_t tb = (size_t)id * DMODEL;
    const size_t pb = (size_t)s * DMODEL;
    float x[4];
    float lsum = 0.0f;
#pragma unroll
    for (int i = 0; i < 4; ++i) {
        const int d = threadIdx.x + (i << 8);
        x[i] = tok[tb + d] + pos[pb + d];
        lsum += x[i];
    }
    const float mean = block_red<false>(lsum, sm) * (1.0f / DMODEL);
    float lv = 0.0f;
#pragma unroll
    for (int i = 0; i < 4; ++i) { const float t = x[i] - mean; lv += t * t; }
    const float rs = rsqrtf(block_red<false>(lv, sm) * (1.0f / DMODEL) + LN_EPS);
    const size_t ob = (size_t)row * DMODEL;
#pragma unroll
    for (int i = 0; i < 4; ++i) {
        const int d = threadIdx.x + (i << 8);
        const float y = (x[i] - mean) * rs * g[d] + bt[d];
        h32[ob + d] = y;
        h16[ob + d] = (half_t)y;
    }
}

// ---- Residual + LayerNorm (in-place safe on base) --------------------------
__global__ __launch_bounds__(256) void add_ln_kernel(
    const float* __restrict__ base, const float* __restrict__ delta,
    const float* __restrict__ g, const float* __restrict__ bt,
    float* __restrict__ h32, half_t* __restrict__ h16) {
    __shared__ float sm[8];
    const size_t ob = (size_t)blockIdx.x * DMODEL;
    float x[4];
    float lsum = 0.0f;
#pragma unroll
    for (int i = 0; i < 4; ++i) {
        const int d = threadIdx.x + (i << 8);
        x[i] = base[ob + d] + delta[ob + d];
        lsum += x[i];
    }
    const float mean = block_red<false>(lsum, sm) * (1.0f / DMODEL);
    float lv = 0.0f;
#pragma unroll
    for (int i = 0; i < 4; ++i) { const float t = x[i] - mean; lv += t * t; }
    const float rs = rsqrtf(block_red<false>(lv, sm) * (1.0f / DMODEL) + LN_EPS);
#pragma unroll
    for (int i = 0; i < 4; ++i) {
        const int d = threadIdx.x + (i << 8);
        const float y = (x[i] - mean) * rs * g[d] + bt[d];
        h32[ob + d] = y;
        h16[ob + d] = (half_t)y;
    }
}

// ---- Fused attention: one (batch, head, 16-query block) per workgroup ------
__global__ __launch_bounds__(128) void attn_kernel(
    const half_t* __restrict__ q, const half_t* __restrict__ k,
    const half_t* __restrict__ v, const float* __restrict__ mask,
    float* __restrict__ ctx) {
    __shared__ float sc[16 * SEQ];   // 64KB; f16 probs alias the front of each row
    __shared__ float rsum[16];
    __shared__ float sm[8];
    const int lane = threadIdx.x & 31;
    const int wave = threadIdx.x >> 5;
    const int qt = blockIdx.x, head = blockIdx.y, b = blockIdx.z;
    const size_t rowbase = (size_t)b * SEQ;
    const int n  = lane & 15;
    const int rb = (lane >> 4) << 3;

    // Phase 1: scores = Q(16x64) @ K^T(64xSEQ), per 16-key tile
    const half_t* Qp = q + (rowbase + qt * 16) * DMODEL + head * DHEAD;
    v16h a0 = frag_A(Qp, DMODEL, lane);
    v16h a1 = frag_A(Qp + 32, DMODEL, lane);
    for (int t = wave; t < SEQ / 16; t += 4) {
        const half_t* Kp = k + (rowbase + t * 16 + n) * DMODEL + head * DHEAD +
                           ((lane >> 4) << 4);
        VH b0, b1;
#pragma unroll
        for (int i = 0; i < 16; ++i) { b0.h[i] = Kp[i]; b1.h[i] = Kp[32 + i]; }
        v8f c = zero8();
        c = wmma_f16(a0, b0.v, c);
        c = wmma_f16(a1, b1.v, c);
        VF cf; cf.v = c;
#pragma unroll
        for (int r = 0; r < 8; ++r) sc[(rb + r) * SEQ + t * 16 + n] = cf.f[r];
    }
    __syncthreads();

    // Phase 2: softmax per row (rows sequential; 128 threads x 8 cols)
    const float scale = 0.125f;  // 1/sqrt(64)
    for (int row = 0; row < 16; ++row) {
        float xv[8];
        float mx = -3.0e38f;
#pragma unroll
        for (int i = 0; i < 8; ++i) {
            const int col = threadIdx.x + (i << 7);
            const float mb = (1.0f - mask[b * SEQ + col]) * -10000.0f;
            xv[i] = sc[row * SEQ + col] * scale + mb;
            mx = fmaxf(mx, xv[i]);
        }
        mx = block_red<true>(mx, sm);    // barrier => all reads of row done
        float lsum = 0.0f;
#pragma unroll
        for (int i = 0; i < 8; ++i) { xv[i] = __expf(xv[i] - mx); lsum += xv[i]; }
        lsum = block_red<false>(lsum, sm);
        half_t* prow = (half_t*)&sc[row * SEQ];  // overwrites only this row
#pragma unroll
        for (int i = 0; i < 8; ++i) prow[threadIdx.x + (i << 7)] = (half_t)xv[i];
        if (threadIdx.x == 0) rsum[row] = lsum;
        __syncthreads();
    }

    // Phase 3: ctx(16x64) = probs(16xSEQ f16, in LDS) @ V(SEQ x 64)
    const int n0 = wave << 4;
    v8f c = zero8();
    const half_t* pbase = (const half_t*)sc;     // row stride = 2*SEQ halves
    const half_t* Vp = v + rowbase * DMODEL + head * DHEAD + n0 + n;
    for (int k0 = 0; k0 < SEQ; k0 += 32) {
        v16h a = frag_A(pbase + k0, 2 * SEQ, lane);
        VH bb;
        const half_t* vp = Vp + (size_t)(k0 + ((lane >> 4) << 4)) * DMODEL;
#pragma unroll
        for (int i = 0; i < 16; ++i) bb.h[i] = vp[(size_t)i * DMODEL];
        c = wmma_f16(a, bb.v, c);
    }
    VF cf; cf.v = c;
#pragma unroll
    for (int r = 0; r < 8; ++r) {
        const int row = rb + r;
        ctx[(rowbase + qt * 16 + row) * DMODEL + head * DHEAD + n0 + n] =
            cf.f[r] / rsum[row];
    }
}

// ---- Host-side orchestration ----------------------------------------------
extern "C" void kernel_launch(void* const* d_in, const int* in_sizes, int n_in,
                              void* d_out, int out_size, void* d_ws, size_t ws_size,
                              hipStream_t stream) {
    const int*   ids  = (const int*)d_in[0];
    const float* mask = (const float*)d_in[1];
    const float* tok  = (const float*)d_in[2];
    const float* pos  = (const float*)d_in[3];
    const float* ln0g = (const float*)d_in[4];
    const float* ln0b = (const float*)d_in[5];
    const float* Wq   = (const float*)d_in[6];
    const float* bq   = (const float*)d_in[7];
    const float* Wk   = (const float*)d_in[8];
    const float* bk   = (const float*)d_in[9];
    const float* Wv   = (const float*)d_in[10];
    const float* bv   = (const float*)d_in[11];
    const float* ln1g = (const float*)d_in[12];
    const float* ln1b = (const float*)d_in[13];
    const float* W1   = (const float*)d_in[14];
    const float* b1   = (const float*)d_in[15];
    const float* W2   = (const float*)d_in[16];
    const float* b2   = (const float*)d_in[17];
    const float* ln2g = (const float*)d_in[18];
    const float* ln2b = (const float*)d_in[19];
    const float* Wlm  = (const float*)d_in[20];
    float* out = (float*)d_out;

    char* w = (char*)d_ws;
    float*  h32   = (float*)w;  w += (size_t)ROWS * DMODEL * 4;
    float*  ctx32 = (float*)w;  w += (size_t)ROWS * DMODEL * 4;
    half_t* h16   = (half_t*)w; w += (size_t)ROWS * DMODEL * 2;
    half_t* q16   = (half_t*)w; w += (size_t)ROWS * DMODEL * 2;
    half_t* k16   = (half_t*)w; w += (size_t)ROWS * DMODEL * 2;
    half_t* v16p  = (half_t*)w; w += (size_t)ROWS * DMODEL * 2;
    half_t* a16   = (half_t*)w; w += (size_t)ROWS * FFDIM * 2;

    embed_ln_kernel<<<ROWS, 256, 0, stream>>>(ids, tok, pos, ln0g, ln0b, h32, h16);

    const dim3 gD(DMODEL / 256, ROWS / 16);   // (4, 128)
    const dim3 gF(FFDIM / 256, ROWS / 16);    // (16, 128)
    const dim3 gV(VOCAB / 256, ROWS / 16);    // (125, 128)
    const dim3 gA(SEQ / 16, NHEAD, BATCH);    // (64, 16, 2)

    for (int l = 0; l < LAYERS; ++l) {
        const float* wq = Wq + (size_t)l * DMODEL * DMODEL;
        const float* wk = Wk + (size_t)l * DMODEL * DMODEL;
        const float* wv = Wv + (size_t)l * DMODEL * DMODEL;
        const float* w1 = W1 + (size_t)l * DMODEL * FFDIM;
        const float* w2 = W2 + (size_t)l * FFDIM * DMODEL;
        const float* bqL = bq + (size_t)l * DMODEL;
        const float* bkL = bk + (size_t)l * DMODEL;
        const float* bvL = bv + (size_t)l * DMODEL;
        const float* b1L = b1 + (size_t)l * FFDIM;
        const float* b2L = b2 + (size_t)l * DMODEL;

        gemm_wmma<false, true><<<gD, 128, 0, stream>>>(h16, wq, bqL, nullptr, q16,
                                                       ROWS, DMODEL, DMODEL);
        gemm_wmma<false, true><<<gD, 128, 0, stream>>>(h16, wk, bkL, nullptr, k16,
                                                       ROWS, DMODEL, DMODEL);
        gemm_wmma<false, true><<<gD, 128, 0, stream>>>(h16, wv, bvL, nullptr, v16p,
                                                       ROWS, DMODEL, DMODEL);
        attn_kernel<<<gA, 128, 0, stream>>>(q16, k16, v16p, mask, ctx32);
        add_ln_kernel<<<ROWS, 256, 0, stream>>>(h32, ctx32,
                                                ln1g + (size_t)l * DMODEL,
                                                ln1b + (size_t)l * DMODEL, h32, h16);
        gemm_wmma<true, true><<<gF, 128, 0, stream>>>(h16, w1, b1L, nullptr, a16,
                                                      ROWS, FFDIM, DMODEL);
        gemm_wmma<false, false><<<gD, 128, 0, stream>>>(a16, w2, b2L, ctx32, nullptr,
                                                        ROWS, DMODEL, FFDIM);
        add_ln_kernel<<<ROWS, 256, 0, stream>>>(h32, ctx32,
                                                ln2g + (size_t)l * DMODEL,
                                                ln2b + (size_t)l * DMODEL, h32, h16);
    }

    gemm_wmma<false, false><<<gV, 128, 0, stream>>>(h16, Wlm, nullptr, out, nullptr,
                                                    ROWS, VOCAB, DMODEL);
}
